// MultiHeadAttention_12111807775102
// MI455X (gfx1250) — matrix operand order repaired and checked by reading, not compiled, not run on hardware
//
#include <hip/hip_runtime.h>
#include <hip/hip_bf16.h>
#include <math.h>

// Problem constants (match reference)
#define Bn 4
#define Tn 2048
#define Dn 512
#define Hn 8
#define DHn 64
#define NEGV (-65504.0f)

typedef __attribute__((ext_vector_type(16))) _Float16 v16h;
typedef __attribute__((ext_vector_type(8)))  _Float16 v8h;
typedef __attribute__((ext_vector_type(8)))  float    v8f;

union H16 { v16h v; v8h h[2]; };

// ---------------------------------------------------------------------------
// Cast f32 -> f16, 4 elements per thread (vectorized b128 load / b64 store).
// ---------------------------------------------------------------------------
__global__ __launch_bounds__(256) void cast_f16_kernel(
    const float* __restrict__ x, _Float16* __restrict__ y, int n4) {
  int i = blockIdx.x * 256 + threadIdx.x;
  if (i >= n4) return;
  const float4 f = ((const float4*)x)[i];
  _Float16* yp = y + (size_t)i * 4;
  yp[0] = (_Float16)f.x;
  yp[1] = (_Float16)f.y;
  yp[2] = (_Float16)f.z;
  yp[3] = (_Float16)f.w;
}

// ---------------------------------------------------------------------------
// Cast + transpose a 512x512 f32 weight matrix into f16 column-major:
//   WT[n*512 + k] = (f16) W[k*512 + n]
// Makes every WMMA B-fragment one contiguous 32-byte load.
// ---------------------------------------------------------------------------
__global__ __launch_bounds__(256) void transpose_w_kernel(
    const float* __restrict__ W, _Float16* __restrict__ WT) {
  int idx = blockIdx.x * 256 + threadIdx.x;
  if (idx >= Dn * Dn) return;
  int n = idx >> 9;
  int k = idx & 511;
  WT[idx] = (_Float16)W[k * Dn + n];
}

// ---------------------------------------------------------------------------
// Epilogue store helper.
// mode 0/1: f16 head-major (B,H,T,DH)   mode 2: f16 (B,H,DH,T)   mode 3: f32
// ---------------------------------------------------------------------------
__device__ __forceinline__ void epi_store(int mode, float y, int r, int c,
                                          _Float16* out16, float* out32) {
  if (mode == 3) {
    out32[(size_t)r * Dn + c] = y;
  } else {
    const int b_ = r >> 11, t = r & (Tn - 1);
    const int h_ = c >> 6,  dh = c & (DHn - 1);
    if (mode == 2)
      out16[(((size_t)(b_ * Hn + h_) * DHn) + dh) * Tn + t] = (_Float16)y;
    else
      out16[(((size_t)(b_ * Hn + h_) * Tn) + t) * DHn + dh] = (_Float16)y;
  }
}

// ---------------------------------------------------------------------------
// WMMA GEMM: Y = X(8192x512, f16) * W(512x512) + bias.
// One wave computes a 32x32 tile (2x2 WMMA sub-tiles). Software-pipelined:
// iteration kb issues loads for kb+32 before its 4 WMMAs so in-flight loads
// cover the wmma latency.
// ---------------------------------------------------------------------------
__global__ __launch_bounds__(32) void gemm_wmma_kernel(
    const _Float16* __restrict__ X, const _Float16* __restrict__ WT,
    const float* __restrict__ bias, _Float16* __restrict__ out16,
    float* __restrict__ out32, int mode) {
  const int lane = threadIdx.x & 31;
  const int m    = lane & 15;
  const int g    = lane >> 4;
  const int off  = g * 16;
  const int rbase = blockIdx.x * 32;
  const int cbase = blockIdx.y * 32;

  v8f acc00 = {}, acc01 = {}, acc10 = {}, acc11 = {};
  const _Float16* arow0 = X  + (size_t)(rbase + m) * Dn;
  const _Float16* arow1 = X  + (size_t)(rbase + 16 + m) * Dn;
  const _Float16* brow0 = WT + (size_t)(cbase + m) * Dn;
  const _Float16* brow1 = WT + (size_t)(cbase + 16 + m) * Dn;

  // prologue: load kb = 0 fragments
  H16 a0, a1; v16h b0, b1;
  a0.h[0] = *(const v8h*)(arow0 + off);
  a0.h[1] = *(const v8h*)(arow0 + 8 + off);
  a1.h[0] = *(const v8h*)(arow1 + off);
  a1.h[1] = *(const v8h*)(arow1 + 8 + off);
  b0 = *(const v16h*)(brow0 + g * 16);
  b1 = *(const v16h*)(brow1 + g * 16);

#pragma unroll 2
  for (int kb = 0; kb < Dn; kb += 32) {
    H16 na0, na1; v16h nb0, nb1;
    const int nk = kb + 32;
    if (nk < Dn) {
      na0.h[0] = *(const v8h*)(arow0 + nk + off);
      na0.h[1] = *(const v8h*)(arow0 + nk + 8 + off);
      na1.h[0] = *(const v8h*)(arow1 + nk + off);
      na1.h[1] = *(const v8h*)(arow1 + nk + 8 + off);
      nb0 = *(const v16h*)(brow0 + nk + g * 16);
      nb1 = *(const v16h*)(brow1 + nk + g * 16);
    } else {
      na0 = a0; na1 = a1; nb0 = b0; nb1 = b1;
    }
    acc00 = __builtin_amdgcn_wmma_f32_16x16x32_f16(false, a0.v, false, b0,
                                                   (short)0, acc00, false, false);
    acc01 = __builtin_amdgcn_wmma_f32_16x16x32_f16(false, a0.v, false, b1,
                                                   (short)0, acc01, false, false);
    acc10 = __builtin_amdgcn_wmma_f32_16x16x32_f16(false, a1.v, false, b0,
                                                   (short)0, acc10, false, false);
    acc11 = __builtin_amdgcn_wmma_f32_16x16x32_f16(false, a1.v, false, b1,
                                                   (short)0, acc11, false, false);
    a0 = na0; a1 = na1; b0 = nb0; b1 = nb1;
  }

  const int   c0  = cbase + m;
  const int   c1  = cbase + 16 + m;
  const float bc0 = bias[c0];
  const float bc1 = bias[c1];
#pragma unroll
  for (int v = 0; v < 8; ++v) {
    const int r0 = rbase + v + g * 8;
    const int r1 = r0 + 16;
    epi_store(mode, acc00[v] + bc0, r0, c0, out16, out32);
    epi_store(mode, acc01[v] + bc1, r0, c1, out16, out32);
    epi_store(mode, acc10[v] + bc0, r1, c0, out16, out32);
    epi_store(mode, acc11[v] + bc1, r1, c1, out16, out32);
  }
}

// ---------------------------------------------------------------------------
// Flash attention with ALiBi + key mask. One wave handles 32 queries (two
// 16-row WMMA tiles) of one (b,h); loops over keys in tiles of 32. Each K/V
// B-fragment is reused by both query tiles -> 16 WMMAs per key tile for the
// same fragment traffic that 8 WMMAs cost before. Online softmax in f32;
// P bounced through a 2KB LDS tile for the C->A relayout. Output written as
// f16 directly into the final-projection input buffer.
// ---------------------------------------------------------------------------
__global__ __launch_bounds__(32) void attn_wmma_kernel(
    const _Float16* __restrict__ Qh, const _Float16* __restrict__ Kh,
    const _Float16* __restrict__ Vt, const int* __restrict__ maskp,
    const float* __restrict__ log_slopes, _Float16* __restrict__ O16) {
  __shared__ _Float16 Pl[32 * 32];

  const int lane = threadIdx.x & 31;
  const int m    = lane & 15;
  const int g    = lane >> 4;
  const int off  = g * 16;

  const int bh = blockIdx.y;
  const int b_ = bh >> 3;
  const int h_ = bh & 7;
  const int qbase = blockIdx.x * 32;

  const _Float16* Q = Qh + (size_t)bh * Tn * DHn;
  const _Float16* K = Kh + (size_t)bh * Tn * DHn;
  const _Float16* V = Vt + (size_t)bh * DHn * Tn;

  const float slope = exp2f(-__expf(log_slopes[h_]));
  const float scale = 0.044194173824159216f;  // 1/sqrt(512)

  // Q A-fragments: 2 query tiles x 2 K-steps (DH=64), hoisted.
  H16 aq[2][2];
#pragma unroll
  for (int qt = 0; qt < 2; ++qt) {
    const _Float16* qrow = Q + (size_t)(qbase + qt * 16 + m) * DHn;
    aq[qt][0].h[0] = *(const v8h*)(qrow + 0  + off);
    aq[qt][0].h[1] = *(const v8h*)(qrow + 8  + off);
    aq[qt][1].h[0] = *(const v8h*)(qrow + 32 + off);
    aq[qt][1].h[1] = *(const v8h*)(qrow + 40 + off);
  }

  v8f oacc[2][4];
  float mr[2][8], lr[2][8];
#pragma unroll
  for (int qt = 0; qt < 2; ++qt) {
#pragma unroll
    for (int d = 0; d < 4; ++d) oacc[qt][d] = {};
#pragma unroll
    for (int v = 0; v < 8; ++v) { mr[qt][v] = -INFINITY; lr[qt][v] = 0.0f; }
  }

  const int* mrow = maskp + b_ * Tn;

  for (int kt = 0; kt < Tn; kt += 32) {
    // Prefetch next key tile's K and V rows into cache (global_prefetch_b8).
    if (kt + 32 < Tn) {
      __builtin_prefetch(K + (size_t)(kt + 32 + m) * DHn, 0, 1);
      __builtin_prefetch(K + (size_t)(kt + 48 + m) * DHn, 0, 1);
      __builtin_prefetch(V + (size_t)(m)      * Tn + kt + 32, 0, 1);
      __builtin_prefetch(V + (size_t)(32 + m) * Tn + kt + 32, 0, 1);
    }

    // ---- K B-fragments for 32 keys (two 16-key N tiles, two K-steps) ----
    const _Float16* kr0 = K + (size_t)(kt + m) * DHn;
    const _Float16* kr1 = K + (size_t)(kt + 16 + m) * DHn;
    v16h bk0a = *(const v16h*)(kr0 + 0  + g * 16);
    v16h bk0b = *(const v16h*)(kr0 + 32 + g * 16);
    v16h bk1a = *(const v16h*)(kr1 + 0  + g * 16);
    v16h bk1b = *(const v16h*)(kr1 + 32 + g * 16);

    // ---- S = Q K^T : 8 WMMAs (2 query tiles x 2 key N-tiles x 2 K-steps)
    v8f s[2][2];
#pragma unroll
    for (int qt = 0; qt < 2; ++qt) {
      v8f t0 = {}, t1 = {};
      t0 = __builtin_amdgcn_wmma_f32_16x16x32_f16(false, aq[qt][0].v, false,
                                                  bk0a, (short)0, t0, false, false);
      t0 = __builtin_amdgcn_wmma_f32_16x16x32_f16(false, aq[qt][1].v, false,
                                                  bk0b, (short)0, t0, false, false);
      t1 = __builtin_amdgcn_wmma_f32_16x16x32_f16(false, aq[qt][0].v, false,
                                                  bk1a, (short)0, t1, false, false);
      t1 = __builtin_amdgcn_wmma_f32_16x16x32_f16(false, aq[qt][1].v, false,
                                                  bk1b, (short)0, t1, false, false);
      s[qt][0] = t0;
      s[qt][1] = t1;
    }

    // ---- scale + ALiBi + key mask, online softmax (per query tile) ----
    const int j0  = kt + m;
    const int j1  = kt + 16 + m;
    const int mk0 = mrow[j0];
    const int mk1 = mrow[j1];
#pragma unroll
    for (int qt = 0; qt < 2; ++qt) {
      float alpha[8];
#pragma unroll
      for (int v = 0; v < 8; ++v) {
        const float iq = (float)(qbase + qt * 16 + v + g * 8);
        float x0 = s[qt][0][v] * scale - fabsf(iq - (float)j0) * slope;
        float x1 = s[qt][1][v] * scale - fabsf(iq - (float)j1) * slope;
        if (mk0 == 0) x0 = NEGV;
        if (mk1 == 0) x1 = NEGV;
        float t = fmaxf(x0, x1);
        t = fmaxf(t, __shfl_xor(t, 1, 32));
        t = fmaxf(t, __shfl_xor(t, 2, 32));
        t = fmaxf(t, __shfl_xor(t, 4, 32));
        t = fmaxf(t, __shfl_xor(t, 8, 32));   // row max (16-lane group)
        const float mn = fmaxf(mr[qt][v], t);
        const float a  = __expf(mr[qt][v] - mn);  // -inf -> 0 on first tile
        const float p0 = __expf(x0 - mn);
        const float p1 = __expf(x1 - mn);
        float rs = p0 + p1;
        rs += __shfl_xor(rs, 1, 32);
        rs += __shfl_xor(rs, 2, 32);
        rs += __shfl_xor(rs, 4, 32);
        rs += __shfl_xor(rs, 8, 32);          // row sum
        lr[qt][v] = lr[qt][v] * a + rs;
        mr[qt][v] = mn;
        alpha[v]  = a;
        s[qt][0][v] = p0;
        s[qt][1][v] = p1;
      }
      // rescale running output accumulators
#pragma unroll
      for (int v = 0; v < 8; ++v) {
        oacc[qt][0][v] *= alpha[v];
        oacc[qt][1][v] *= alpha[v];
        oacc[qt][2][v] *= alpha[v];
        oacc[qt][3][v] *= alpha[v];
      }
      // P (f16) into LDS: C-layout -> row-major
#pragma unroll
      for (int v = 0; v < 8; ++v) {
        const int row = qt * 16 + v + g * 8;
        Pl[row * 32 + m]      = (_Float16)s[qt][0][v];
        Pl[row * 32 + 16 + m] = (_Float16)s[qt][1][v];
      }
    }
    __syncthreads();
    H16 ap0, ap1;
    ap0.h[0] = *(const v8h*)(Pl + m * 32 + off);
    ap0.h[1] = *(const v8h*)(Pl + m * 32 + 8 + off);
    ap1.h[0] = *(const v8h*)(Pl + (16 + m) * 32 + off);
    ap1.h[1] = *(const v8h*)(Pl + (16 + m) * 32 + 8 + off);
    __syncthreads();

    // ---- O += P * V : 8 WMMAs, V fragments shared by both query tiles ----
    {
      const size_t kc = (size_t)kt + g * 16;
      v16h bv0 = *(const v16h*)(V + (size_t)(0  + m) * Tn + kc);
      v16h bv1 = *(const v16h*)(V + (size_t)(16 + m) * Tn + kc);
      v16h bv2 = *(const v16h*)(V + (size_t)(32 + m) * Tn + kc);
      v16h bv3 = *(const v16h*)(V + (size_t)(48 + m) * Tn + kc);
      oacc[0][0] = __builtin_amdgcn_wmma_f32_16x16x32_f16(false, ap0.v, false, bv0,
                                                          (short)0, oacc[0][0], false, false);
      oacc[0][1] = __builtin_amdgcn_wmma_f32_16x16x32_f16(false, ap0.v, false, bv1,
                                                          (short)0, oacc[0][1], false, false);
      oacc[0][2] = __builtin_amdgcn_wmma_f32_16x16x32_f16(false, ap0.v, false, bv2,
                                                          (short)0, oacc[0][2], false, false);
      oacc[0][3] = __builtin_amdgcn_wmma_f32_16x16x32_f16(false, ap0.v, false, bv3,
                                                          (short)0, oacc[0][3], false, false);
      oacc[1][0] = __builtin_amdgcn_wmma_f32_16x16x32_f16(false, ap1.v, false, bv0,
                                                          (short)0, oacc[1][0], false, false);
      oacc[1][1] = __builtin_amdgcn_wmma_f32_16x16x32_f16(false, ap1.v, false, bv1,
                                                          (short)0, oacc[1][1], false, false);
      oacc[1][2] = __builtin_amdgcn_wmma_f32_16x16x32_f16(false, ap1.v, false, bv2,
                                                          (short)0, oacc[1][2], false, false);
      oacc[1][3] = __builtin_amdgcn_wmma_f32_16x16x32_f16(false, ap1.v, false, bv3,
                                                          (short)0, oacc[1][3], false, false);
    }
  }

  // ---- normalize and write O as f16 in (B,T,D) layout ----
#pragma unroll
  for (int qt = 0; qt < 2; ++qt) {
#pragma unroll
    for (int v = 0; v < 8; ++v) {
      const float inv = 1.0f / lr[qt][v];
      const int   t   = qbase + qt * 16 + v + g * 8;
      _Float16* op = O16 + ((size_t)(b_ * Tn + t)) * Dn + h_ * DHn + m;
      op[0]  = (_Float16)(oacc[qt][0][v] * inv);
      op[16] = (_Float16)(oacc[qt][1][v] * inv);
      op[32] = (_Float16)(oacc[qt][2][v] * inv);
      op[48] = (_Float16)(oacc[qt][3][v] * inv);
    }
  }
}

// ---------------------------------------------------------------------------
// Host-side orchestration
// ---------------------------------------------------------------------------
extern "C" void kernel_launch(void* const* d_in, const int* in_sizes, int n_in,
                              void* d_out, int out_size, void* d_ws, size_t ws_size,
                              hipStream_t stream) {
  (void)in_sizes; (void)n_in; (void)out_size; (void)ws_size;
  const float* q  = (const float*)d_in[0];
  const float* k  = (const float*)d_in[1];
  const float* v  = (const float*)d_in[2];
  const int*   mk = (const int*)d_in[3];
  const float* Wq = (const float*)d_in[4];
  const float* bq = (const float*)d_in[5];
  const float* Wk = (const float*)d_in[6];
  const float* bk = (const float*)d_in[7];
  const float* Wv = (const float*)d_in[8];
  const float* bv = (const float*)d_in[9];
  const float* Wo = (const float*)d_in[10];
  const float* bo = (const float*)d_in[11];
  const float* ls = (const float*)d_in[12];
  float* out = (float*)d_out;

  // Workspace layout (bytes): 4x f16 transposed weights (2 MiB),
  // 6x f16 (8192x512) tensors (48 MiB). O16 aliases Xq (dead after Q-proj).
  char* ws = (char*)d_ws;
  const size_t szWT = (size_t)Dn * Dn * sizeof(_Float16);             // 512 KB
  const size_t szX  = (size_t)Bn * Tn * Dn * sizeof(_Float16);        // 8 MiB
  _Float16* WqT = (_Float16*)(ws);
  _Float16* WkT = (_Float16*)(ws + szWT);
  _Float16* WvT = (_Float16*)(ws + 2 * szWT);
  _Float16* WoT = (_Float16*)(ws + 3 * szWT);
  _Float16* Xq  = (_Float16*)(ws + 4 * szWT);
  _Float16* Xk  = (_Float16*)(ws + 4 * szWT + szX);
  _Float16* Xv  = (_Float16*)(ws + 4 * szWT + 2 * szX);
  _Float16* Qh  = (_Float16*)(ws + 4 * szWT + 3 * szX);
  _Float16* Kh  = (_Float16*)(ws + 4 * szWT + 4 * szX);
  _Float16* Vt  = (_Float16*)(ws + 4 * szWT + 5 * szX);
  _Float16* O16 = Xq;  // alias: Xq is dead once the Q projection has run

  // 1) cast activations to f16 (once), cast+transpose weights to f16
  const int n4 = (Bn * Tn * Dn) / 4;
  cast_f16_kernel<<<(n4 + 255) / 256, 256, 0, stream>>>(q, Xq, n4);
  cast_f16_kernel<<<(n4 + 255) / 256, 256, 0, stream>>>(k, Xk, n4);
  cast_f16_kernel<<<(n4 + 255) / 256, 256, 0, stream>>>(v, Xv, n4);
  const int twBlocks = (Dn * Dn) / 256;
  transpose_w_kernel<<<twBlocks, 256, 0, stream>>>(Wq, WqT);
  transpose_w_kernel<<<twBlocks, 256, 0, stream>>>(Wk, WkT);
  transpose_w_kernel<<<twBlocks, 256, 0, stream>>>(Wv, WvT);
  transpose_w_kernel<<<twBlocks, 256, 0, stream>>>(Wo, WoT);

  // 2) Q/K/V projections (8192x512 @ 512x512) on WMMA, 32x32 tile per wave
  dim3 ggrid((Bn * Tn) / 32, Dn / 32);
  gemm_wmma_kernel<<<ggrid, 32, 0, stream>>>(Xq, WqT, bq, Qh, nullptr, 0);
  gemm_wmma_kernel<<<ggrid, 32, 0, stream>>>(Xk, WkT, bk, Kh, nullptr, 1);
  gemm_wmma_kernel<<<ggrid, 32, 0, stream>>>(Xv, WvT, bv, Vt, nullptr, 2);

  // 3) flash attention (ALiBi + mask), one wave per 32 queries per (b,h)
  attn_wmma_kernel<<<dim3(Tn / 32, Bn * Hn), 32, 0, stream>>>(
      Qh, Kh, Vt, mk, ls, O16);

  // 4) output projection -> f32 d_out
  gemm_wmma_kernel<<<ggrid, 32, 0, stream>>>(O16, WoT, bo, nullptr, out, 3);
}
